// VAE_32667521253538
// MI455X (gfx1250) — matrix operand order repaired and checked by reading, not compiled, not run on hardware
//
#include <hip/hip_runtime.h>
#include <hip/hip_bf16.h>

typedef __attribute__((ext_vector_type(16))) _Float16 v16h;
typedef __attribute__((ext_vector_type(8)))  float    v8f;

static inline unsigned cdiv_u(long a, long b) { return (unsigned)((a + b - 1) / b); }

// ---------------- elementwise / graph kernels ----------------

__global__ void k_zero_f32(float* __restrict__ p, long n) {
  long i = (long)blockIdx.x * blockDim.x + threadIdx.x;
  if (i < n) p[i] = 0.0f;
}

__global__ void k_deg(const int* __restrict__ dst, int E, float* __restrict__ deg) {
  int e = blockIdx.x * blockDim.x + threadIdx.x;
  if (e < E) atomicAdd(&deg[dst[e]], 1.0f);
}

__global__ void k_dinv(float* __restrict__ d, int N) {
  int i = blockIdx.x * blockDim.x + threadIdx.x;
  if (i < N) d[i] = rsqrtf(d[i] + 1.0f);   // +1 self-loop; deg >= 1 guaranteed
}

// out[i,:] = tbl[x[i],:]   (float4-vectorized row gather)
__global__ void k_gather_rows(const int* __restrict__ x, const float* __restrict__ tbl,
                              float* __restrict__ out, long n4, int c4) {
  long g = (long)blockIdx.x * blockDim.x + threadIdx.x;
  if (g >= n4) return;
  long i = g / c4;
  long c = g - i * c4;
  ((float4*)out)[g] = ((const float4*)tbl)[(long)x[i] * c4 + c];
}

// out[i,:] = in[i,:] * dinv[i]^2   (self-loop contribution)
__global__ void k_agg_init(const float* __restrict__ in, const float* __restrict__ dinv,
                           float* __restrict__ out, long n4, int c4) {
  long g = (long)blockIdx.x * blockDim.x + threadIdx.x;
  if (g >= n4) return;
  long i = g / c4;
  float s = dinv[i]; s *= s;
  float4 v = ((const float4*)in)[g];
  v.x *= s; v.y *= s; v.z *= s; v.w *= s;
  ((float4*)out)[g] = v;
}

// one block (blockDim == C) per edge: out[dst,:] += in[src,:]*dinv[src]*dinv[dst]
__global__ void k_agg_edge(const int* __restrict__ src, const int* __restrict__ dst,
                           const float* __restrict__ dinv, const float* __restrict__ in,
                           float* __restrict__ out, int E, int C) {
  int e = blockIdx.x;
  if (e >= E) return;
  int s = src[e], d = dst[e];
  float nrm = dinv[s] * dinv[d];
  int c = threadIdx.x;
  atomicAdd(&out[(long)d * C + c], in[(long)s * C + c] * nrm);
}

__global__ void k_bias_relu(float* __restrict__ h, const float* __restrict__ b, long n, int C) {
  long g = (long)blockIdx.x * blockDim.x + threadIdx.x;
  if (g >= n) return;
  int c = (int)(g % C);
  float v = h[g] + b[c];
  h[g] = v > 0.0f ? v : 0.0f;
}

// z = mu + eps * exp(0.5*logvar)
__global__ void k_z(const float* __restrict__ mu, const float* __restrict__ lv,
                    const float* __restrict__ eps, float* __restrict__ z, long n) {
  long i = (long)blockIdx.x * blockDim.x + threadIdx.x;
  if (i < n) z[i] = mu[i] + eps[i] * __expf(0.5f * lv[i]);
}

// capitalize = sigmoid(z @ Wcap + bcap), Wcap is [L,1]
__global__ void k_cap(const float* __restrict__ z, const float* __restrict__ w,
                      const float* __restrict__ b, float* __restrict__ out, int N, int L) {
  int i = blockIdx.x * blockDim.x + threadIdx.x;
  if (i >= N) return;
  float acc = b[0];
  const float* zr = z + (long)i * L;
  for (int k = 0; k < L; ++k) acc += zr[k] * w[k];
  out[i] = 1.0f / (1.0f + __expf(-acc));
}

// ---------------- WMMA GEMM path ----------------
// Pack fp32 W[K x M] (row-major) into f16 WMMA-B operand tiles.
// pid = ((kt*(M/16)+nt)*32 + lane)*16 + j
//   maps to W[k = kt*32 + j + 16*(lane>=16)][n = nt*16 + lane%16]
__global__ void k_pack_b(const float* __restrict__ W, _Float16* __restrict__ Wp, int K, int M) {
  long pid = (long)blockIdx.x * blockDim.x + threadIdx.x;
  long total = (long)K * M;
  if (pid >= total) return;
  int j    = (int)(pid & 15);
  int lane = (int)((pid >> 4) & 31);
  long t   = pid >> 9;
  int ntm  = M >> 4;
  int nt   = (int)(t % ntm);
  int kt   = (int)(t / ntm);
  int k = kt * 32 + j + ((lane & 16) ? 16 : 0);
  int n = nt * 16 + (lane & 15);
  Wp[pid] = (_Float16)W[(long)k * M + n];
}

// C[N x M] = act(A[N x K] * Bpacked + bias), f32 A converted to f16 on the fly,
// f32 accumulate via v_wmma_f32_16x16x32_f16. One wave per 16x16 C tile.
// Requires K % 32 == 0. ldc = row stride of C in floats.
__global__ void k_gemm_wmma(const float* __restrict__ A, const _Float16* __restrict__ Bp,
                            const float* __restrict__ bias, float* __restrict__ C,
                            int N, int K, int M, int ldc, int act) {
  int wid  = (int)(((long)blockIdx.x * blockDim.x + threadIdx.x) >> 5);
  int lane = threadIdx.x & 31;
  int ntm = M >> 4;
  int ntn = (N + 15) >> 4;
  if (wid >= ntn * ntm) return;           // wave-uniform: EXEC all-ones at WMMA
  int rt = wid / ntm, ct = wid % ntm;
  int row0 = rt << 4, col0 = ct << 4;
  int m  = lane & 15;
  int hi = lane >> 4;                     // 0: lanes 0-15, 1: lanes 16-31
  int ar = row0 + m; if (ar >= N) ar = N - 1;   // clamped dup row; stores guarded
  const float*    arow = A  + (long)ar * K;
  const _Float16* bpt  = Bp + ((long)ct * 32 + lane) * 16;
  long bstep = (long)ntm * 512;           // halves per K-tile of packed B
  v8f acc = {};
  int kiters = K >> 5;
  for (int kt = 0; kt < kiters; ++kt) {
    const float4* ap = (const float4*)(arow + (kt << 5));
    // A operand layout: lanes<16 halves = K{0..15}; lanes>=16 = K{16..31}
    float4 f0 = ap[hi * 4 + 0];
    float4 f1 = ap[hi * 4 + 1];
    float4 f2 = ap[hi * 4 + 2];
    float4 f3 = ap[hi * 4 + 3];
    v16h a;
    a[0]=(_Float16)f0.x;  a[1]=(_Float16)f0.y;  a[2]=(_Float16)f0.z;  a[3]=(_Float16)f0.w;
    a[4]=(_Float16)f1.x;  a[5]=(_Float16)f1.y;  a[6]=(_Float16)f1.z;  a[7]=(_Float16)f1.w;
    a[8]=(_Float16)f2.x;  a[9]=(_Float16)f2.y;  a[10]=(_Float16)f2.z; a[11]=(_Float16)f2.w;
    a[12]=(_Float16)f3.x; a[13]=(_Float16)f3.y; a[14]=(_Float16)f3.z; a[15]=(_Float16)f3.w;
    v16h b = *(const v16h*)(bpt + (long)kt * bstep);
    acc = __builtin_amdgcn_wmma_f32_16x16x32_f16(
        /*neg_a=*/false, a, /*neg_b=*/false, b,
        /*c_mod=*/(short)0, acc, /*reuse_a=*/false, /*reuse_b=*/false);
  }
  // D layout: lane holds C[row0 + 8*hi + v][col0 + lane%16] in acc[v]
  int n  = lane & 15;
  int mb = hi * 8;
  float bv = bias ? bias[col0 + n] : 0.0f;
#pragma unroll
  for (int v = 0; v < 8; ++v) {
    int r = row0 + mb + v;
    if (r < N) {
      float val = acc[v] + bv;
      if (act) val = fmaxf(val, 0.0f);
      C[(long)r * ldc + col0 + n] = val;
    }
  }
}

// ---------------- orchestration ----------------

extern "C" void kernel_launch(void* const* d_in, const int* in_sizes, int n_in,
                              void* d_out, int out_size, void* d_ws, size_t ws_size,
                              hipStream_t stream) {
  const int*   x    = (const int*)d_in[0];
  const int*   edge = (const int*)d_in[1];
  const float* eps  = (const float*)d_in[2];
  const float* emb  = (const float*)d_in[3];
  const float* W1   = (const float*)d_in[4];
  const float* b1   = (const float*)d_in[5];
  const float* W2   = (const float*)d_in[6];
  const float* b2   = (const float*)d_in[7];
  const float* Wmu  = (const float*)d_in[8];
  const float* bmu  = (const float*)d_in[9];
  const float* Wlv  = (const float*)d_in[10];
  const float* blv  = (const float*)d_in[11];
  const float* W3   = (const float*)d_in[12];
  const float* b3   = (const float*)d_in[13];
  const float* W4   = (const float*)d_in[14];
  const float* b4   = (const float*)d_in[15];
  const float* Wcap = (const float*)d_in[16];
  const float* bcap = (const float*)d_in[17];

  const int N = in_sizes[0];
  const int E = in_sizes[1] / 2;
  const int H = in_sizes[5];          // 256
  const int L = in_sizes[9];          // 64
  const int F = in_sizes[4] / H;      // 128
  const int V = in_sizes[14] / H;     // 128

  const int* srcIdx = edge;
  const int* dstIdx = edge + E;

  // workspace carve-out (256B aligned)
  char* w = (char*)d_ws;
  auto carve = [&](size_t bytes) { char* p = w; w += (bytes + 255) & ~(size_t)255; return p; };
  float*    dinv  = (float*)carve(sizeof(float) * (size_t)N);
  float*    bufA  = (float*)carve(sizeof(float) * (size_t)N * H);
  float*    bufB  = (float*)carve(sizeof(float) * (size_t)N * H);
  float*    embW1 = (float*)carve(sizeof(float) * (size_t)V * H);
  _Float16* W1p   = (_Float16*)carve(sizeof(_Float16) * (size_t)F * H);
  _Float16* W2p   = (_Float16*)carve(sizeof(_Float16) * (size_t)H * H);
  _Float16* Wmup  = (_Float16*)carve(sizeof(_Float16) * (size_t)H * L);
  _Float16* Wlvp  = (_Float16*)carve(sizeof(_Float16) * (size_t)H * L);
  _Float16* W3p   = (_Float16*)carve(sizeof(_Float16) * (size_t)L * H);
  _Float16* W4p   = (_Float16*)carve(sizeof(_Float16) * (size_t)H * V);
  float* zbuf = bufA;   // [N,L], reuses bufA after XW2 is consumed

  // output layout: recon_x [N,V] | capitalize [N] | mu [N,L] | logvar [N,L]
  float* out_recon = (float*)d_out;
  float* out_cap   = out_recon + (size_t)N * V;
  float* out_mu    = out_cap + N;
  float* out_lv    = out_mu + (size_t)N * L;

  const int TB = 256;
  auto pack = [&](const float* Wm, _Float16* Wp, int K, int M) {
    k_pack_b<<<cdiv_u((long)K * M, TB), TB, 0, stream>>>(Wm, Wp, K, M);
  };
  auto gemm = [&](const float* A, const _Float16* Bp, const float* bias, float* Cc,
                  int n_, int k_, int m_, int ldc, int act) {
    long waves = (long)((n_ + 15) / 16) * (m_ / 16);
    k_gemm_wmma<<<cdiv_u(waves * 32, TB), TB, 0, stream>>>(A, Bp, bias, Cc, n_, k_, m_, ldc, act);
  };

  // degree -> dinv
  k_zero_f32<<<cdiv_u(N, TB), TB, 0, stream>>>(dinv, (long)N);
  k_deg<<<cdiv_u(E, TB), TB, 0, stream>>>(dstIdx, E, dinv);
  k_dinv<<<cdiv_u(N, TB), TB, 0, stream>>>(dinv, N);

  // pack all weights to f16 WMMA-B layout (tiny, done every call for determinism)
  pack(W1, W1p, F, H); pack(W2, W2p, H, H);
  pack(Wmu, Wmup, H, L); pack(Wlv, Wlvp, H, L);
  pack(W3, W3p, L, H); pack(W4, W4p, H, V);

  // layer 1: fold embedding into W1:  (emb@W1)[x]  (V=128 rows -> tiny GEMM + gather)
  gemm(emb, W1p, nullptr, embW1, V, F, H, H, 0);
  k_gather_rows<<<cdiv_u((long)N * (H / 4), TB), TB, 0, stream>>>(x, embW1, bufA, (long)N * (H / 4), H / 4);
  k_agg_init<<<cdiv_u((long)N * (H / 4), TB), TB, 0, stream>>>(bufA, dinv, bufB, (long)N * (H / 4), H / 4);
  k_agg_edge<<<E, H, 0, stream>>>(srcIdx, dstIdx, dinv, bufA, bufB, E, H);
  k_bias_relu<<<cdiv_u((long)N * H, TB), TB, 0, stream>>>(bufB, b1, (long)N * H, H);   // h1

  // layer 2
  gemm(bufB, W2p, nullptr, bufA, N, H, H, H, 0);                                       // XW2
  k_agg_init<<<cdiv_u((long)N * (H / 4), TB), TB, 0, stream>>>(bufA, dinv, bufB, (long)N * (H / 4), H / 4);
  k_agg_edge<<<E, H, 0, stream>>>(srcIdx, dstIdx, dinv, bufA, bufB, E, H);
  k_bias_relu<<<cdiv_u((long)N * H, TB), TB, 0, stream>>>(bufB, b2, (long)N * H, H);   // h2

  // heads: mu / logvar straight into d_out, then z
  gemm(bufB, Wmup, bmu, out_mu, N, H, L, L, 0);
  gemm(bufB, Wlvp, blv, out_lv, N, H, L, L, 0);
  k_z<<<cdiv_u((long)N * L, TB), TB, 0, stream>>>(out_mu, out_lv, eps, zbuf, (long)N * L);

  // decoder
  gemm(zbuf, W3p, b3, bufB, N, L, H, H, 1);          // t = relu(z@W3+b3)
  gemm(bufB, W4p, b4, out_recon, N, H, V, V, 0);     // recon_x
  k_cap<<<cdiv_u(N, TB), TB, 0, stream>>>(zbuf, Wcap, bcap, out_cap, N, L);
}
